// MaskedAttentionHead_46445776338971
// MI455X (gfx1250) — matrix operand order repaired and checked by reading, not compiled, not run on hardware
//
#include <hip/hip_runtime.h>

typedef _Float16 half_t;
typedef __attribute__((ext_vector_type(16))) _Float16 v16h;
typedef __attribute__((ext_vector_type(8)))  _Float16 v8h;
typedef __attribute__((ext_vector_type(8)))  float    v8f;
typedef __attribute__((ext_vector_type(4)))  float    f32x4;   // native clang vector
typedef __attribute__((ext_vector_type(4)))  unsigned int u32x4;
typedef __attribute__((ext_vector_type(8)))  unsigned int u32x8;

constexpr int B_ = 512;   // batches
constexpr int S_ = 256;   // sequence
constexpr int E_ = 384;   // embedding
constexpr int H_ = 64;    // head size

constexpr int WSTR = 392;  // padded Wt row stride (halves): 784B = 16B-aligned, 4-bank stagger
constexpr int QSTR = 72;   // LDS row stride (halves) for Q/K
constexpr int VSTR = 264;  // LDS row stride for V^T
constexpr int PSTR = 40;   // per-wave P staging stride

__device__ __forceinline__ v8f wmma_f16(v16h a, v16h b, v8f c) {
  // D = A(16x32 f16) * B(32x16 f16) + C(16x16 f32)
  return __builtin_amdgcn_wmma_f32_16x16x32_f16(false, a, false, b, (short)0, c,
                                                false, false);
}

__device__ __forceinline__ v8f vzero8() {
  v8f z;
#pragma unroll
  for (int i = 0; i < 8; ++i) z[i] = 0.0f;
  return z;
}

// A fragment (16x32, f16) from a row-major f16 buffer.
// Layout: lanes 0-15 hold K=[0..7],[16..23]; lanes 16-31 hold K=[8..15],[24..31].
__device__ __forceinline__ v16h load_a_f16(const half_t* base, int ld, int m0,
                                           int k0, int lane) {
  const half_t* p = base + (m0 + (lane & 15)) * ld + k0 + ((lane >> 4) << 3);
  v8h lo = *(const v8h*)(p);
  v8h hi = *(const v8h*)(p + 16);
  v16h a;
#pragma unroll
  for (int j = 0; j < 8; ++j) { a[j] = lo[j]; a[8 + j] = hi[j]; }
  return a;
}

// A fragment from a row-major f32 global buffer (streaming: non-temporal).
__device__ __forceinline__ v16h load_a_f32(const float* base, int ld, int m0,
                                           int k0, int lane) {
  const float* p =
      base + (size_t)(m0 + (lane & 15)) * ld + k0 + ((lane >> 4) << 3);
  f32x4 a0 = __builtin_nontemporal_load((const f32x4*)(p));
  f32x4 a1 = __builtin_nontemporal_load((const f32x4*)(p + 4));
  f32x4 b0 = __builtin_nontemporal_load((const f32x4*)(p + 16));
  f32x4 b1 = __builtin_nontemporal_load((const f32x4*)(p + 20));
  v16h a;
#pragma unroll
  for (int j = 0; j < 4; ++j) {
    a[j]      = (half_t)a0[j];
    a[4 + j]  = (half_t)a1[j];
    a[8 + j]  = (half_t)b0[j];
    a[12 + j] = (half_t)b1[j];
  }
  return a;
}

// B fragment (32x16, f16). `base` row n holds B's column n along K.
// Layout: lanes 0-15 hold K=[0..7],[16..23]; lanes 16-31 hold K=[8..15],[24..31].
__device__ __forceinline__ v16h load_b_f16(const half_t* base, int ld, int n0,
                                           int k0, int lane) {
  const half_t* p = base + (n0 + (lane & 15)) * ld + k0 + ((lane >> 4) << 3);
  v8h lo = *(const v8h*)(p);
  v8h hi = *(const v8h*)(p + 16);
  v16h b;
#pragma unroll
  for (int j = 0; j < 8; ++j) { b[j] = lo[j]; b[8 + j] = hi[j]; }
  return b;
}

// Convert Wq/Wk/Wv [E,H] fp32 -> transposed padded [3][H][WSTR] f16 in workspace.
__global__ void wt_prep_kernel(const float* __restrict__ Wq,
                               const float* __restrict__ Wk,
                               const float* __restrict__ Wv,
                               half_t* __restrict__ Wt) {
  int idx = blockIdx.x * blockDim.x + threadIdx.x;
  if (idx >= 3 * H_ * E_) return;
  int mat = idx / (H_ * E_);
  int rem = idx - mat * (H_ * E_);
  int h = rem / E_;
  int e = rem - h * E_;
  const float* W = (mat == 0) ? Wq : ((mat == 1) ? Wk : Wv);
  Wt[(mat * H_ + h) * WSTR + e] = (half_t)W[e * H_ + h];
}

__global__ __launch_bounds__(256, 1)
void attn_kernel(const float* __restrict__ x, const half_t* __restrict__ Wt,
                 float* __restrict__ out) {
  __shared__ __align__(16) half_t WtL[3 * H_ * WSTR];   // weights, TDM-loaded
  __shared__ __align__(16) half_t Qs[S_ * QSTR];        // Q row-major f16
  __shared__ __align__(16) half_t Ks[S_ * QSTR];        // K row-major f16
  __shared__ __align__(16) half_t Vt[H_ * VSTR];        // V transposed [H][S]
  __shared__ __align__(16) half_t Ps[8 * 16 * PSTR];    // per-wave P staging

  const int lane = threadIdx.x & 31;
  const int w = threadIdx.x >> 5;       // wave id 0..7
  const int b = blockIdx.x;             // batch
  const float* xb = x + (size_t)b * S_ * E_;

  const int Nc = lane & 15;             // column index in C layout
  const int moL = (lane >> 4) << 3;     // lanes>=16 own rows 8..15

  // ---- TDM: DMA the f16 weight block (3*64 x 392 halves) global -> LDS ----
  if (w == 0) {
    unsigned ldsOff = (unsigned)(size_t)(&WtL[0]);     // LDS aperture: offset in [31:0]
    unsigned long long ga = (unsigned long long)(size_t)Wt;
    constexpr unsigned ROWS = 3 * H_;                  // 192
    constexpr unsigned D1S = (unsigned)WSTR * ROWS;    // dim1 stride (elements)
    u32x4 g0;
    g0[0] = 1u;                                        // count=1 (valid user D#)
    g0[1] = ldsOff;                                    // lds_addr
    g0[2] = (unsigned)(ga & 0xFFFFFFFFull);            // global_addr[31:0]
    g0[3] = (unsigned)((ga >> 32) & 0x01FFFFFFull) | (2u << 30);  // [56:32] | type=2
    u32x8 g1;
    g1[0] = 0x00010000u;                 // workgroup_mask=0, data_size=1 (2B)
    g1[1] = ((unsigned)WSTR) << 16;      // tensor_dim0[15:0]
    g1[2] = ROWS << 16;                  // tensor_dim0[31:16]=0 | tensor_dim1[15:0]
    g1[3] = ((unsigned)WSTR) << 16;      // tensor_dim1[31:16]=0 | tile_dim0
    g1[4] = ROWS;                        // tile_dim1 (tile_dim2=0)
    g1[5] = (unsigned)WSTR;              // tensor_dim0_stride[31:0]
    g1[6] = (D1S & 0xFFFFu) << 16;       // stride0[47:32]=0 | dim1_stride[15:0]
    g1[7] = D1S >> 16;                   // dim1_stride[47:16]
    asm volatile("tensor_load_to_lds %0, %1" ::"s"(g0), "s"(g1) : "memory");
    __builtin_amdgcn_s_wait_tensorcnt(0);
  }
  __syncthreads();

  // ---------------- Phase 1: Q,K,V = x[b] @ W (f16 WMMA, f32 acc) ----------
  for (int mt = 0; mt < 2; ++mt) {
    const int m0 = (w * 2 + mt) * 16;
    v8f acc[12];
#pragma unroll
    for (int i = 0; i < 12; ++i) acc[i] = vzero8();

    for (int k = 0; k < E_ / 32; ++k) {           // 12 K-steps of 32
      v16h ax = load_a_f32(xb, E_, m0, k * 32, lane);
#pragma unroll
      for (int mat = 0; mat < 3; ++mat) {
#pragma unroll
        for (int n = 0; n < 4; ++n) {
          v16h bw = load_b_f16(WtL + mat * H_ * WSTR, WSTR, n * 16, k * 32, lane);
          acc[mat * 4 + n] = wmma_f16(ax, bw, acc[mat * 4 + n]);
        }
      }
    }
    const int mo = m0 + moL;
#pragma unroll
    for (int n = 0; n < 4; ++n) {
#pragma unroll
      for (int r = 0; r < 8; ++r) {
        Qs[(mo + r) * QSTR + n * 16 + Nc] = (half_t)acc[n][r];
        Ks[(mo + r) * QSTR + n * 16 + Nc] = (half_t)acc[4 + n][r];
        Vt[(n * 16 + Nc) * VSTR + mo + r] = (half_t)acc[8 + n][r];  // transpose
      }
    }
  }
  __syncthreads();

  // ---------------- Phase 2: scores, causal softmax, O = P @ V -------------
  half_t* Pw = Ps + w * 16 * PSTR;
  for (int qt = 0; qt < 2; ++qt) {
    const int qi = w * 2 + qt;          // query tile 0..15 (wave-uniform)
    const int q0 = qi * 16;
    v16h aq0 = load_a_f16(Qs, QSTR, q0, 0, lane);
    v16h aq1 = load_a_f16(Qs, QSTR, q0, 32, lane);

    v8f sc[16];
#pragma unroll
    for (int j = 0; j < 16; ++j) {
      if (j <= qi) {                    // causal: skip fully-masked key tiles
        v8f s = vzero8();
        s = wmma_f16(aq0, load_b_f16(Ks, QSTR, j * 16, 0, lane), s);
        s = wmma_f16(aq1, load_b_f16(Ks, QSTR, j * 16, 32, lane), s);
#pragma unroll
        for (int r = 0; r < 8; ++r) {
          float v = s[r] * 0.125f;      // 1/sqrt(64)
          if (j == qi) v = (Nc <= moL + r) ? v : -1.0e30f;  // diagonal mask
          s[r] = v;
        }
        sc[j] = s;
      } else {
        v8f s;
#pragma unroll
        for (int r = 0; r < 8; ++r) s[r] = -1.0e30f;
        sc[j] = s;
      }
    }

    // Row softmax: row (moL + r) lives across the 16 lanes of a half-wave.
    float rmax[8], rsum[8], inv[8];
#pragma unroll
    for (int r = 0; r < 8; ++r) {
      float m = -1.0e30f;
#pragma unroll
      for (int j = 0; j < 16; ++j) m = fmaxf(m, sc[j][r]);
#pragma unroll
      for (int d = 1; d < 16; d <<= 1) m = fmaxf(m, __shfl_xor(m, d, 32));
      rmax[r] = m;
      rsum[r] = 0.0f;
    }
#pragma unroll
    for (int j = 0; j < 16; ++j) {
#pragma unroll
      for (int r = 0; r < 8; ++r) {
        float e = (j <= qi) ? __expf(sc[j][r] - rmax[r]) : 0.0f;
        sc[j][r] = e;
        rsum[r] += e;
      }
    }
#pragma unroll
    for (int r = 0; r < 8; ++r) {
      float s = rsum[r];
#pragma unroll
      for (int d = 1; d < 16; d <<= 1) s += __shfl_xor(s, d, 32);
      inv[r] = 1.0f / s;
    }

    // O accumulation: stage P tile-pairs through LDS (C-layout -> A-fragment).
    v8f o[4];
#pragma unroll
    for (int n = 0; n < 4; ++n) o[n] = vzero8();
#pragma unroll
    for (int s = 0; s < 8; ++s) {
      if (2 * s <= qi) {                // only non-masked K range
#pragma unroll
        for (int r = 0; r < 8; ++r) {
          Pw[(moL + r) * PSTR + 0 + Nc]  = (half_t)sc[2 * s][r];
          Pw[(moL + r) * PSTR + 16 + Nc] = (half_t)sc[2 * s + 1][r];
        }
        v16h ap = load_a_f16(Pw, PSTR, 0, 0, lane);
#pragma unroll
        for (int n = 0; n < 4; ++n)
          o[n] = wmma_f16(ap, load_b_f16(Vt, VSTR, n * 16, s * 32, lane), o[n]);
      }
    }
#pragma unroll
    for (int n = 0; n < 4; ++n)
#pragma unroll
      for (int r = 0; r < 8; ++r)
        __builtin_nontemporal_store(
            o[n][r] * inv[r],
            &out[((size_t)b * S_ + q0 + moL + r) * H_ + n * 16 + Nc]);
  }
}

extern "C" void kernel_launch(void* const* d_in, const int* in_sizes, int n_in,
                              void* d_out, int out_size, void* d_ws,
                              size_t ws_size, hipStream_t stream) {
  (void)in_sizes; (void)n_in; (void)out_size; (void)ws_size;
  const float* x  = (const float*)d_in[0];
  const float* Wq = (const float*)d_in[1];
  const float* Wk = (const float*)d_in[2];
  const float* Wv = (const float*)d_in[3];
  float* out = (float*)d_out;
  half_t* Wt = (half_t*)d_ws;  // needs 3*64*392*2 = 150528 bytes

  const int wtN = 3 * H_ * E_;
  wt_prep_kernel<<<(wtN + 255) / 256, 256, 0, stream>>>(Wq, Wk, Wv, Wt);
  attn_kernel<<<B_, 256, 0, stream>>>(x, Wt, out);
}
